// GNNRecommender_58729382805523
// MI455X (gfx1250) — matrix operand order repaired and checked by reading, not compiled, not run on hardware
//
#include <hip/hip_runtime.h>

typedef __attribute__((ext_vector_type(16))) _Float16 v16h;
typedef __attribute__((ext_vector_type(8)))  float    v8f;

#define GCN_N 100000
#define GCN_F 16   // hidden/output feature dim

// ---------------------------------------------------------------------------
// Degree / normalization kernels
// ---------------------------------------------------------------------------
__global__ void k_deg_init(float* __restrict__ deg, int n) {
  int i = blockIdx.x * blockDim.x + threadIdx.x;
  if (i < n) deg[i] = 1.0f;  // self-loop contributes 1 to every node's degree
}

__global__ void k_deg_edges(const int* __restrict__ dst, float* __restrict__ deg, int e) {
  int i = blockIdx.x * blockDim.x + threadIdx.x;
  if (i < e) atomicAdd(&deg[dst[i]], 1.0f);
}

__global__ void k_dinv(float* __restrict__ deg, int n) {
  int i = blockIdx.x * blockDim.x + threadIdx.x;
  if (i < n) {
    float d = deg[i];
    deg[i] = (d > 0.0f) ? rsqrtf(d) : 0.0f;  // in place: deg -> d^-1/2
  }
}

// ---------------------------------------------------------------------------
// WMMA GEMM: H[n,16] = (RELU ? relu(A) : A)[n,K] @ W[K,16]
// One wave computes one 16x16 output tile. K is a COMPILE-TIME constant so
// all fragment bounds checks fold away (K=128 -> 4 dense chunks; K=16 -> one
// chunk with the upper half compile-time zero). f16 inputs, f32 accumulate.
// Layouts per CDNA5 ISA 7.12.2:
//   A (16x32, f16): lane<16 -> row M=lane; khalf = 8*(lane>=16);
//                   VGPR v: K = 16*(v>>2) + khalf + 2*(v&3) (+0,+1)
//   B (32x16, f16): lane -> col N=lane%16; VGPR v: K = 16*(v>>2) + khalf + 2*(v&3) (+0,+1), same as A
//   C/D (16x16, f32): lane -> N=lane%16; VGPR r: M = r + 8*(lane>=16)
// ---------------------------------------------------------------------------
template <int K, bool RELU>
__global__ void k_gemm_wmma(const float* __restrict__ A, const float* __restrict__ W,
                            float* __restrict__ H, int n) {
  const int lane  = threadIdx.x & 31;
  const int wave  = blockIdx.x * (blockDim.x >> 5) + (threadIdx.x >> 5);
  const int ntile = n >> 4;
  if (wave >= ntile) return;  // per-wave uniform: EXEC stays all-ones

  const int row0   = wave << 4;
  const int rA     = row0 + (lane & 15);       // A row this lane feeds
  const int colB   = lane & 15;                // B/D column this lane feeds
  const int khalfA = (lane >> 4) << 3;         // 0 or 8
  const int khalfB = (lane >> 4) << 4;         // 0 or 16
  const float* __restrict__ Arow = A + (size_t)rA * K;

  v8f acc = {};
#pragma unroll
  for (int kc = 0; kc < K; kc += 32) {
    v16h a = {};
    v16h b = {};
#pragma unroll
    for (int v = 0; v < 8; ++v) {
      // ---- A fragment: ka, ka+1 are consecutive & 8B-aligned -> one b64 load
      const int ka = kc + ((v >> 2) << 4) + khalfA + ((v & 3) << 1);
      if (ka + 16 <= K || ka < K) {  // folds at compile time (ka even, K%16==0)
        float2 p = *(const float2*)(Arow + ka);
        if (RELU) { p.x = fmaxf(p.x, 0.0f); p.y = fmaxf(p.y, 0.0f); }
        a[2 * v]     = (_Float16)p.x;
        a[2 * v + 1] = (_Float16)p.y;
      }
      // ---- B fragment: consecutive K -> rows 16 floats apart in W
      const int kb = kc + ((v >> 2) << 4) + khalfA + ((v & 3) << 1);
      if (kb < K) {
        b[2 * v]     = (_Float16)W[(size_t)kb * GCN_F + colB];
        b[2 * v + 1] = (_Float16)W[(size_t)(kb + 1) * GCN_F + colB];
      }
    }
    acc = __builtin_amdgcn_wmma_f32_16x16x32_f16(
        /*neg_a=*/false, a, /*neg_b=*/false, b,
        /*c_mod=*/(short)0, acc, /*reuse_a=*/false, /*reuse_b=*/false);
  }

  const int rowBase = row0 + ((lane >> 4) << 3);
#pragma unroll
  for (int r = 0; r < 8; ++r)
    H[(size_t)(rowBase + r) * GCN_F + colB] = acc[r];
}

// ---------------------------------------------------------------------------
// out[i,f] = bias[f] + dinv[i]^2 * h[i,f]   (self-loop term + bias; also
// serves as the full initialization of the destination buffer)
// ---------------------------------------------------------------------------
__global__ void k_aggr_init(const float* __restrict__ h, const float* __restrict__ dinv,
                            const float* __restrict__ bias, float* __restrict__ out, int n) {
  int t = blockIdx.x * blockDim.x + threadIdx.x;
  if (t < n * GCN_F) {
    int i = t >> 4, f = t & 15;
    float di = dinv[i];
    out[t] = bias[f] + di * di * h[t];
  }
}

// ---------------------------------------------------------------------------
// Edge scatter: 16 consecutive lanes handle one edge -> the h[src] gather and
// the atomic burst into out[dst] are each a single contiguous 64B transaction.
// out[dst,f] += dinv[src]*dinv[dst] * h[src,f]
// ---------------------------------------------------------------------------
__global__ void k_aggr_edges(const float* __restrict__ h, const float* __restrict__ dinv,
                             const int* __restrict__ src, const int* __restrict__ dst,
                             float* __restrict__ out, int e) {
  long long t = (long long)blockIdx.x * blockDim.x + threadIdx.x;
  if (t < (long long)e * GCN_F) {
    int ei = (int)(t >> 4);
    int f  = (int)(t & 15);
    int s  = src[ei];
    int d  = dst[ei];
    float norm = dinv[s] * dinv[d];
    atomicAdd(&out[(size_t)d * GCN_F + f], norm * h[(size_t)s * GCN_F + f]);
  }
}

// ---------------------------------------------------------------------------
// Launch
// ---------------------------------------------------------------------------
extern "C" void kernel_launch(void* const* d_in, const int* in_sizes, int n_in,
                              void* d_out, int out_size, void* d_ws, size_t ws_size,
                              hipStream_t stream) {
  const float* x    = (const float*)d_in[0];
  const int*   edge = (const int*)d_in[1];   // [2, E] flattened
  const float* W1   = (const float*)d_in[2];
  const float* b1   = (const float*)d_in[3];
  const float* W2   = (const float*)d_in[4];
  const float* b2   = (const float*)d_in[5];
  float* out = (float*)d_out;

  const int n = GCN_N;
  const int e = in_sizes[1] / 2;
  const int* src = edge;
  const int* dst = edge + e;

  // workspace layout (floats): dinv[n] | h1[n*16] | a1[n*16] | h2[n*16]
  float* ws   = (float*)d_ws;
  float* dinv = ws;
  float* h1   = dinv + n;
  float* a1   = h1 + (size_t)n * GCN_F;
  float* h2   = a1 + (size_t)n * GCN_F;

  const int B = 256;
  const int gN  = (n + B - 1) / B;
  const int gE  = (e + B - 1) / B;
  const int gNF = (n * GCN_F + B - 1) / B;
  const int gEF = (int)(((long long)e * GCN_F + B - 1) / B);
  const int gemmBlocks = ((n >> 4) + 7) / 8;   // 8 waves (256 threads) per block

  // normalization (shared by both layers)
  k_deg_init <<<gN, B, 0, stream>>>(dinv, n);
  k_deg_edges<<<gE, B, 0, stream>>>(dst, dinv, e);
  k_dinv     <<<gN, B, 0, stream>>>(dinv, n);

  // layer 1: h1 = x @ W1 ; a1 = segment_sum(norm * h1) + b1
  k_gemm_wmma<128, false><<<gemmBlocks, 256, 0, stream>>>(x, W1, h1, n);
  k_aggr_init <<<gNF, B, 0, stream>>>(h1, dinv, b1, a1, n);
  k_aggr_edges<<<gEF, B, 0, stream>>>(h1, dinv, src, dst, a1, e);

  // layer 2: h2 = relu(a1) @ W2 ; out = segment_sum(norm * h2) + b2
  k_gemm_wmma<16, true><<<gemmBlocks, 256, 0, stream>>>(a1, W2, h2, n);
  k_aggr_init <<<gNF, B, 0, stream>>>(h2, dinv, b2, out, n);
  k_aggr_edges<<<gEF, B, 0, stream>>>(h2, dinv, src, dst, out, e);
}
